// Arcface_10771777978861
// MI455X (gfx1250) — compile-verified, matrix operand order checked by reading, not run on hardware
//
#include <hip/hip_runtime.h>
#include <hip/hip_bf16.h>

typedef _Float16 half4 __attribute__((ext_vector_type(4)));
typedef _Float16 half8 __attribute__((ext_vector_type(8)));
typedef _Float16 v16h  __attribute__((ext_vector_type(16)));
typedef float    v8f   __attribute__((ext_vector_type(8)));

#define B_ 8
#define S_ 5
#define W_ 1024
#define H_ 128
#define P_ 10
#define LDS_STRIDE 136   // 128 + 8 halves = 272B row pitch (16B aligned, bank-spread)

__device__ const int c_pi[P_] = {0,0,0,0,1,1,1,2,2,3};
__device__ const int c_pj[P_] = {1,2,3,4,2,3,4,3,4,4};

// ---------------------------------------------------------------------------
// Kernel 1: L2-normalize each 128-float row, emit f16. One wave32 per row,
// each lane owns 4 contiguous floats; cross-lane reduce via shfl_xor.
// ---------------------------------------------------------------------------
__global__ __launch_bounds__(256) void k_normalize(const float* __restrict__ x,
                                                   _Float16* __restrict__ xn) {
    const int lane = threadIdx.x & 31;
    const int wid  = threadIdx.x >> 5;
    const int row  = blockIdx.x * 8 + wid;                 // 0 .. B*S*W-1
    const float4 v = *(const float4*)(x + (size_t)row * H_ + lane * 4);
    float s = v.x * v.x + v.y * v.y + v.z * v.z + v.w * v.w;
#pragma unroll
    for (int off = 16; off > 0; off >>= 1) s += __shfl_xor(s, off, 32);
    const float rn = rsqrtf(s);
    half4 o;
    o.x = (_Float16)(v.x * rn);
    o.y = (_Float16)(v.y * rn);
    o.z = (_Float16)(v.z * rn);
    o.w = (_Float16)(v.w * rn);
    *(half4*)(xn + (size_t)row * H_ + lane * 4) = o;
}

// ---------------------------------------------------------------------------
// Kernel 2: scores[bp] = 5 * Xn[b,pi] @ Xn[b,pj]^T   (1024x1024x128, f16 WMMA)
// Block = 256 threads = 8 waves, computes a 128x128 C tile; full K=128 of both
// operand tiles staged in LDS once. Wave (waveM,waveN) owns a 32x64 slab:
// 2 (M) x 4 (N) accumulators of 16x16, K-loop of 4 x (k=32) WMMA steps.
// ---------------------------------------------------------------------------
__global__ __launch_bounds__(256) void k_gemm(const _Float16* __restrict__ xn,
                                              float* __restrict__ scores) {
    __shared__ _Float16 sA[128 * LDS_STRIDE];
    __shared__ _Float16 sB[128 * LDS_STRIDE];

    const int bp = blockIdx.z;
    const int b  = bp / P_;
    const int p  = bp - b * P_;
    const int m0 = blockIdx.y * 128;
    const int n0 = blockIdx.x * 128;

    const _Float16* __restrict__ Abase = xn + ((size_t)(b * S_ + c_pi[p]) * W_ + m0) * H_;
    const _Float16* __restrict__ Bbase = xn + ((size_t)(b * S_ + c_pj[p]) * W_ + n0) * H_;

    // Cooperative tile load: 2048 16-byte chunks per tile, 8 per thread.
    {
        const int t = threadIdx.x;
#pragma unroll
        for (int it = 0; it < 8; ++it) {
            const int cid = t + it * 256;
            const int row = cid >> 4;
            const int cc  = (cid & 15) * 8;   // halves
            *(half8*)(sA + row * LDS_STRIDE + cc) = *(const half8*)(Abase + row * H_ + cc);
            *(half8*)(sB + row * LDS_STRIDE + cc) = *(const half8*)(Bbase + row * H_ + cc);
        }
    }
    __syncthreads();

    const int lane  = threadIdx.x & 31;
    const int wid   = threadIdx.x >> 5;
    const int waveM = wid >> 1;   // 0..3 -> 32-row slab
    const int waveN = wid & 1;    // 0..1 -> 64-col slab
    const int r     = lane & 15;
    const int h     = lane >> 4;

    const v8f zero = {0.f, 0.f, 0.f, 0.f, 0.f, 0.f, 0.f, 0.f};
    v8f acc[2][4];
#pragma unroll
    for (int mt = 0; mt < 2; ++mt)
#pragma unroll
        for (int nt = 0; nt < 4; ++nt) acc[mt][nt] = zero;

#pragma unroll
    for (int ks = 0; ks < 4; ++ks) {
        const int kb = ks * 32;
        v16h afrag[2], bfrag[4];
        // A-fragment: lane L -> row L%16 of the 16-row tile, chunks at
        // k = kb + h*8 and kb + 16 + h*8 (documented 16-bit A layout).
#pragma unroll
        for (int mt = 0; mt < 2; ++mt) {
            const int rowa = waveM * 32 + mt * 16 + r;
            const _Float16* pa = sA + rowa * LDS_STRIDE + kb + h * 8;
            half8 lo = *(const half8*)pa;
            half8 hi = *(const half8*)(pa + 16);
            afrag[mt] = __builtin_shufflevector(lo, hi, 0,1,2,3,4,5,6,7,
                                                        8,9,10,11,12,13,14,15);
        }
        // B-fragment: C = A*B^T means WMMA-B column n is row n of Xn[b,pj]
        // -> identical per-lane pattern as A.
#pragma unroll
        for (int nt = 0; nt < 4; ++nt) {
            const int rowb = waveN * 64 + nt * 16 + r;
            const _Float16* pb = sB + rowb * LDS_STRIDE + kb + h * 8;
            half8 lo = *(const half8*)pb;
            half8 hi = *(const half8*)(pb + 16);
            bfrag[nt] = __builtin_shufflevector(lo, hi, 0,1,2,3,4,5,6,7,
                                                        8,9,10,11,12,13,14,15);
        }
#pragma unroll
        for (int mt = 0; mt < 2; ++mt)
#pragma unroll
            for (int nt = 0; nt < 4; ++nt)
                acc[mt][nt] = __builtin_amdgcn_wmma_f32_16x16x32_f16(
                    false, afrag[mt], false, bfrag[nt],
                    (short)0, acc[mt][nt], false, false);
    }

    // Epilogue: C layout -> lane L, vgpr d holds (m = d + 8*(L>>4), n = L&15).
    float* __restrict__ out = scores + (size_t)bp * ((size_t)W_ * W_);
#pragma unroll
    for (int mt = 0; mt < 2; ++mt) {
        const int mbase = m0 + waveM * 32 + mt * 16 + h * 8;
#pragma unroll
        for (int nt = 0; nt < 4; ++nt) {
            const int nidx = n0 + waveN * 64 + nt * 16 + r;
#pragma unroll
            for (int d = 0; d < 8; ++d)
                out[(size_t)(mbase + d) * W_ + nidx] = acc[mt][nt][d] * 5.0f;
        }
    }
}

// ---------------------------------------------------------------------------
// Kernel 3: ret_target — first v with target[b,pj,v] == target[b,pi,w], else 0
// (argmax over a 0/1 match vector == first match index).
// ---------------------------------------------------------------------------
__global__ __launch_bounds__(256) void k_targets(const int* __restrict__ tgt,
                                                 float* __restrict__ out) {
    const int idx = blockIdx.x * 256 + threadIdx.x;   // 0 .. B*P*W-1
    const int w   = idx & (W_ - 1);
    const int bp  = idx >> 10;
    const int b   = bp / P_;
    const int p   = bp - b * P_;
    const int t0  = tgt[(b * S_ + c_pi[p]) * W_ + w];
    const int* __restrict__ t1 = tgt + (b * S_ + c_pj[p]) * W_;
    int ret = 0;
    for (int v = 0; v < W_; ++v) {
        if (t1[v] == t0) { ret = v; break; }
    }
    out[idx] = (float)ret;
}

// ---------------------------------------------------------------------------
extern "C" void kernel_launch(void* const* d_in, const int* in_sizes, int n_in,
                              void* d_out, int out_size, void* d_ws, size_t ws_size,
                              hipStream_t stream) {
    (void)in_sizes; (void)n_in; (void)out_size; (void)ws_size;
    const float* x   = (const float*)d_in[0];
    const int*   tgt = (const int*)d_in[1];
    float*       out = (float*)d_out;
    _Float16*    xn  = (_Float16*)d_ws;   // needs B*S*W*H*2 = 10.5 MB

    // 1) normalize f32 -> f16 (one wave per row, 8 rows per block)
    k_normalize<<<(B_ * S_ * W_) / 8, 256, 0, stream>>>(x, xn);

    // 2) 80 x (1024x1024x128) WMMA GEMMs, 128x128 tile per block
    dim3 g(W_ / 128, W_ / 128, B_ * P_);
    k_gemm<<<g, 256, 0, stream>>>(xn, out);

    // 3) match-index targets, appended after the 80M score floats
    float* tout = out + (size_t)B_ * P_ * W_ * W_;
    k_targets<<<(B_ * P_ * W_) / 256, 256, 0, stream>>>(tgt, tout);
}
